// CausalSelfAttention_74921409511712
// MI455X (gfx1250) — matrix operand order repaired and checked by reading, not compiled, not run on hardware
//
#include <hip/hip_runtime.h>

// ---------------------------------------------------------------------------
// CausalSelfAttention forward for MI455X (gfx1250, wave32, WMMA 16x16x32 f16)
// B=2, T=4096, C=768, H=12, D=64
// Flash attention computed in TRANSPOSED form (S^T = K Q^T, O^T = V^T P^T):
// every WMMA operand comes straight from a naturally-laid-out LDS tile, the
// softmax needs a single cross-half shuffle per tile, and P / O move as
// b128 stores.  Async global->LDS K/V tiles, double buffered on ASYNCcnt.
// ---------------------------------------------------------------------------

typedef __attribute__((ext_vector_type(16))) _Float16 v16h;
typedef __attribute__((ext_vector_type(8)))  _Float16 v8h;
typedef __attribute__((ext_vector_type(8)))  float    v8f;
typedef __attribute__((ext_vector_type(4)))  int      v4i;

static constexpr int kB = 2;
static constexpr int kT = 4096;
static constexpr int kC = 768;
static constexpr int kH = 12;
static constexpr int kD = 64;
static constexpr int k3C = 3 * kC;          // 2304

#ifndef __has_builtin
#define __has_builtin(x) 0
#endif

#if __has_builtin(__builtin_amdgcn_global_load_async_to_lds_b128)
#define HAVE_ASYNC_LDS 1
#else
#define HAVE_ASYNC_LDS 0
#endif

// 16-byte global -> LDS copy. Async (ASYNCcnt) when the gfx1250 builtin is
// available; falls back to a plain load+store otherwise.
__device__ __forceinline__ void async_b128(const _Float16* g, _Float16* l) {
#if HAVE_ASYNC_LDS
  __builtin_amdgcn_global_load_async_to_lds_b128((v4i*)g, (v4i*)l,
                                                 /*offset=*/0, /*cpol=*/0);
#else
  *(v8h*)l = *(const v8h*)g;
#endif
}

template <int N>
__device__ __forceinline__ void wait_asynccnt() {
#if HAVE_ASYNC_LDS
#if __has_builtin(__builtin_amdgcn_s_wait_asynccnt)
  __builtin_amdgcn_s_wait_asynccnt(N);
#else
  asm volatile("s_wait_asynccnt %0" ::"i"(N) : "memory");
#endif
#endif
}

__device__ __forceinline__ v8f wmma_f16(v16h a, v16h b, v8f c) {
  // D = A(16x32) x B(32x16) + C(16x16), f32 accum
  return __builtin_amdgcn_wmma_f32_16x16x32_f16(
      /*neg_a=*/false, a, /*neg_b=*/false, b,
      /*c_mod=*/(short)0, c, /*reuse_a=*/false, /*reuse_b=*/false);
}

// A-fragment (16x32 f16): lane m = lane&15, K chunks [8*half,+8) and [16+8*half,+8)
__device__ __forceinline__ v16h frag_a(const _Float16* p0, int ld, int lane) {
  const _Float16* p = p0 + (lane & 15) * ld + ((lane >> 4) << 3);
  v8h lo = *(const v8h*)p;
  v8h hi = *(const v8h*)(p + 16);
  return __builtin_shufflevector(lo, hi, 0, 1, 2, 3, 4, 5, 6, 7,
                                 8, 9, 10, 11, 12, 13, 14, 15);
}

// B-fragment (32x16 f16) from an LDS tile stored [n][k]: lane n = lane&15,
// K chunks [8*half,+8) and [16+8*half,+8).
__device__ __forceinline__ v16h frag_b(const _Float16* p0, int ld, int lane) {
  const _Float16* p = p0 + (lane & 15) * ld + ((lane >> 4) << 3);
  v8h lo = *(const v8h*)p;
  v8h hi = *(const v8h*)(p + 16);
  return __builtin_shufflevector(lo, hi, 0, 1, 2, 3, 4, 5, 6, 7,
                                 8, 9, 10, 11, 12, 13, 14, 15);
}

// Stage a 32(k) x 64(n) f32 weight tile transposed into LDS [n][k] as f16,
// two k-rows packed per 32-bit LDS store.  128 threads.
__device__ __forceinline__ void stage_w_tile(const float* __restrict__ W,
                                             int ldw, int k0, int n0,
                                             _Float16 (*Bs)[32], int tid) {
  const int kp2 = (tid & 15) * 2;   // k pair 0,2,..,30
  const int nb  = (tid >> 4) * 8;   // n sub-block of 8
  const float* s0 = W + (size_t)(k0 + kp2) * ldw + n0 + nb;
  const float* s1 = s0 + ldw;
  float4 a0 = ((const float4*)s0)[0], a1 = ((const float4*)s0)[1];
  float4 b0 = ((const float4*)s1)[0], b1 = ((const float4*)s1)[1];
  float r0[8] = {a0.x, a0.y, a0.z, a0.w, a1.x, a1.y, a1.z, a1.w};
  float r1[8] = {b0.x, b0.y, b0.z, b0.w, b1.x, b1.y, b1.z, b1.w};
#pragma unroll
  for (int i = 0; i < 8; ++i) {
    union { _Float16 h[2]; unsigned int u; } p;
    p.h[0] = (_Float16)r0[i];
    p.h[1] = (_Float16)r1[i];
    *(unsigned int*)&Bs[nb + i][kp2] = p.u;
  }
}

// ---------------------------------------------------------------------------
// Kernel 1: QKV projection.  qkv = x @ Wqkv + bqkv -> f16 scratch.
// Q,K laid out [B,H,T,D]; V laid out transposed [B,H,D,T] (b128 stores);
// Q pre-scaled by 1/sqrt(D).  Tile 64x64, K-step 32, 4 waves/block.
// grid = (3C/64, (B*T)/64) = (36, 128), block = 128.
// ---------------------------------------------------------------------------
__global__ __launch_bounds__(128) void qkv_proj_kernel(
    const float* __restrict__ x, const float* __restrict__ Wqkv,
    const float* __restrict__ bqkv,
    _Float16* __restrict__ Qh, _Float16* __restrict__ Kh,
    _Float16* __restrict__ Vh) {
  __shared__ _Float16 As[64][32];   // [m][k]
  __shared__ _Float16 Bs[64][32];   // [n][k]

  const int tid = threadIdx.x;
  const int lane = tid & 31, wave = tid >> 5;
  const int wm = wave >> 1, wn = wave & 1;
  const int n0 = blockIdx.x * 64;
  const int m0 = blockIdx.y * 64;

  const int arow = tid >> 1, ac0 = (tid & 1) * 16;

  v8f acc[2][2] = {};

  for (int kt = 0; kt < kC / 32; ++kt) {
    const int k0 = kt * 32;
    {  // stage x tile (f32 -> f16), row-major [m][k]
      const float* src = x + (size_t)(m0 + arow) * kC + k0 + ac0;
      _Float16 tmp[16];
#pragma unroll
      for (int i = 0; i < 4; ++i) {
        float4 v = ((const float4*)src)[i];
        tmp[4 * i + 0] = (_Float16)v.x; tmp[4 * i + 1] = (_Float16)v.y;
        tmp[4 * i + 2] = (_Float16)v.z; tmp[4 * i + 3] = (_Float16)v.w;
      }
      *(v8h*)&As[arow][ac0]     = *(const v8h*)&tmp[0];
      *(v8h*)&As[arow][ac0 + 8] = *(const v8h*)&tmp[8];
    }
    stage_w_tile(Wqkv, k3C, k0, n0, Bs, tid);
    __syncthreads();

    v16h a0 = frag_a(&As[wm * 32 + 0][0], 32, lane);
    v16h a1 = frag_a(&As[wm * 32 + 16][0], 32, lane);
    v16h b0 = frag_b(&Bs[wn * 32 + 0][0], 32, lane);
    v16h b1 = frag_b(&Bs[wn * 32 + 16][0], 32, lane);
    acc[0][0] = wmma_f16(a0, b0, acc[0][0]);
    acc[0][1] = wmma_f16(a0, b1, acc[0][1]);
    acc[1][0] = wmma_f16(a1, b0, acc[1][0]);
    acc[1][1] = wmma_f16(a1, b1, acc[1][1]);
    __syncthreads();
  }

  // epilogue: bias (+ Q scale), scatter. 64 | 768 so (q|k|v, head) uniform.
  const int three = n0 / kC;
  const int head  = (n0 % kC) / kD;
  const int half8 = (lane >> 4) << 3;
  if (three == 2) {
    // V: [B,H,D,T]; r runs along t -> one b128 store per 16x16 tile.
#pragma unroll
    for (int im = 0; im < 2; ++im)
#pragma unroll
      for (int in = 0; in < 2; ++in) {
        const int d = wn * 32 + in * 16 + (lane & 15);
        const float bias = bqkv[n0 + d];
        const int gm0 = m0 + wm * 32 + im * 16 + half8;  // 8-aligned
        const int bb = gm0 >> 12;
        const int tt0 = gm0 & 4095;
        _Float16 tmp[8];
#pragma unroll
        for (int r = 0; r < 8; ++r) tmp[r] = (_Float16)(acc[im][in][r] + bias);
        *(v8h*)&Vh[((size_t)((bb * kH + head) * kD + d)) * kT + tt0] =
            *(const v8h*)tmp;
      }
  } else {
    const float scl = (three == 0) ? 0.125f : 1.0f;  // fold 1/sqrt(D) into Q
    _Float16* dst = (three == 0) ? Qh : Kh;
#pragma unroll
    for (int im = 0; im < 2; ++im)
#pragma unroll
      for (int in = 0; in < 2; ++in) {
        const int d = wn * 32 + in * 16 + (lane & 15);
        const float bias = bqkv[n0 + d];
#pragma unroll
        for (int r = 0; r < 8; ++r) {
          const int gm = m0 + wm * 32 + im * 16 + half8 + r;
          const int bb = gm >> 12;
          const int tt = gm & 4095;
          dst[((size_t)((bb * kH + head) * kT + tt)) * kD + d] =
              (_Float16)((acc[im][in][r] + bias) * scl);
        }
      }
  }
}

// ---------------------------------------------------------------------------
// Kernel 2: causal flash attention, transposed formulation.
// Per block: 64 q rows x one (head,batch).  Wave w owns q columns
// [w*16, w*16+16); lane owns one q column (lane&15) across both halves.
//   S^T(64key x 16q) = K(64x64d) @ Q^T : A=Ks[key][d], B=Qs[q][d]
//   O^T(64d  x 16q) = V^T(64dx64k) @ P^T: A=Vt[d][key], B=Ps[q][key]
// grid = (T/64, H, B), block = 128.
// ---------------------------------------------------------------------------
__device__ __forceinline__ void issue_kv_tile(const _Float16* __restrict__ kbase,
                                              const _Float16* __restrict__ vtbase,
                                              int kt, _Float16 (*Ksb)[64],
                                              _Float16 (*Vtb)[64], int tid) {
  const int row = tid >> 1, c0 = (tid & 1) * 32;
  const _Float16* ksrc = kbase + (size_t)(kt * 64 + row) * kD + c0;   // [key][d]
  const _Float16* vsrc = vtbase + (size_t)row * kT + kt * 64 + c0;    // [d][key]
#pragma unroll
  for (int i = 0; i < 4; ++i) {
    async_b128(ksrc + 8 * i, &Ksb[row][c0 + 8 * i]);
    async_b128(vsrc + 8 * i, &Vtb[row][c0 + 8 * i]);
  }
}

__global__ __launch_bounds__(128) void attn_kernel(
    const _Float16* __restrict__ Qh, const _Float16* __restrict__ Kh,
    const _Float16* __restrict__ Vh, _Float16* __restrict__ Oh) {
  __shared__ _Float16 Qs[64][64];        // [q][d]
  __shared__ _Float16 Ks[2][64][64];     // [key][d]  double buffered
  __shared__ _Float16 Vt[2][64][64];     // [d][key]  double buffered
  __shared__ _Float16 Ps[4][16][64];     // per-wave P^T staging [q][key]

  const int tid = threadIdx.x, lane = tid & 31, w = tid >> 5;
  const int qt = blockIdx.x, head = blockIdx.y, b = blockIdx.z;
  const _Float16* kbase  = Kh + (size_t)(b * kH + head) * kT * kD;  // [t][d]
  const _Float16* vtbase = Vh + (size_t)(b * kH + head) * kD * kT;  // [d][t]
  const int half8 = (lane >> 4) << 3;
  const int qcol  = lane & 15;           // this lane's q column (both halves)

  {  // Q tile (async, 4 ops/wave)
    const int row = tid >> 1, c0 = (tid & 1) * 32;
    const _Float16* src =
        Qh + (size_t)(b * kH + head) * kT * kD + (size_t)(qt * 64 + row) * kD + c0;
#pragma unroll
    for (int i = 0; i < 4; ++i) async_b128(src + 8 * i, &Qs[row][c0 + 8 * i]);
  }
  issue_kv_tile(kbase, vtbase, 0, Ks[0], Vt[0], tid);   // 8 ops/wave

  float mrun = -1e30f, lsum = 0.f;       // per-lane (per q column) stats
  v8f o[4] = {};                         // O^T strip: rows d, col = qcol

  for (int kt = 0; kt <= qt; ++kt) {
    const int cur = kt & 1;
    if (kt < qt) {
      issue_kv_tile(kbase, vtbase, kt + 1, Ks[cur ^ 1], Vt[cur ^ 1], tid);
      wait_asynccnt<8>();   // retire tile kt (+Q), keep tile kt+1 in flight
    } else {
      wait_asynccnt<0>();
    }
    __syncthreads();        // tile kt visible to all waves

    // ---- S^T strip (64key x 16q): s[ct] rows = key ct*16+half8+r ----
    v8f s[4] = {};
#pragma unroll
    for (int ks = 0; ks < kD; ks += 32) {
      v16h bq = frag_b(&Qs[w * 16][ks], 64, lane);
#pragma unroll
      for (int ct = 0; ct < 4; ++ct) {
        v16h ak = frag_a(&Ks[cur][ct * 16][ks], 64, lane);
        s[ct] = wmma_f16(ak, bq, s[ct]);
      }
    }

    // ---- causal mask (diagonal tile only): mask key > q ----
    if (kt == qt) {
      const int qrel = w * 16 + qcol;
#pragma unroll
      for (int ct = 0; ct < 4; ++ct)
#pragma unroll
        for (int r = 0; r < 8; ++r)
          if (ct * 16 + half8 + r > qrel) s[ct][r] = -1e30f;
    }

    // ---- online softmax: per-lane column reduction + one shuffle ----
    float m = -1e30f;
#pragma unroll
    for (int ct = 0; ct < 4; ++ct)
#pragma unroll
      for (int r = 0; r < 8; ++r) m = fmaxf(m, s[ct][r]);
    m = fmaxf(m, __shfl_xor(m, 16, 32));
    const float mn = fmaxf(mrun, m);
    const float alpha = __expf(mrun - mn);
    mrun = mn;
    float rs = 0.f;
#pragma unroll
    for (int ct = 0; ct < 4; ++ct)
#pragma unroll
      for (int r = 0; r < 8; ++r) {
        const float e = __expf(s[ct][r] - mn);
        s[ct][r] = e;
        rs += e;
      }
    rs += __shfl_xor(rs, 16, 32);
    lsum = lsum * alpha + rs;
#pragma unroll
    for (int dt = 0; dt < 4; ++dt) o[dt] *= alpha;

    // ---- stage P^T: lane's 8 keys are consecutive -> b128 stores ----
#pragma unroll
    for (int ct = 0; ct < 4; ++ct) {
      _Float16 tmp[8];
#pragma unroll
      for (int r = 0; r < 8; ++r) tmp[r] = (_Float16)s[ct][r];
      *(v8h*)&Ps[w][qcol][ct * 16 + half8] = *(const v8h*)tmp;
    }

    // ---- O^T strip += V^T @ P^T ----
#pragma unroll
    for (int ks = 0; ks < 64; ks += 32) {
      v16h bp = frag_b(&Ps[w][0][ks], 64, lane);
#pragma unroll
      for (int dt = 0; dt < 4; ++dt) {
        v16h av = frag_a(&Vt[cur][dt * 16][ks], 64, lane);
        o[dt] = wmma_f16(av, bp, o[dt]);
      }
    }
    __syncthreads();   // compute done before buffer cur is overwritten (kt+2)
  }

  // ---- normalize, write O as [B,T,C] f16: lane's 8 d's contiguous ----
  const float rl = 1.0f / lsum;
  const int tt = qt * 64 + w * 16 + qcol;
#pragma unroll
  for (int dt = 0; dt < 4; ++dt) {
    _Float16 tmp[8];
#pragma unroll
    for (int r = 0; r < 8; ++r) tmp[r] = (_Float16)(o[dt][r] * rl);
    *(v8h*)&Oh[((size_t)(b * kT + tt)) * kC + head * kD + dt * 16 + half8] =
        *(const v8h*)tmp;
  }
}

// ---------------------------------------------------------------------------
// Kernel 3: output projection.  out = Oh(f16) @ Wout + bout  (f32 out)
// grid = (C/64, (B*T)/64) = (12, 128), block = 128.
// ---------------------------------------------------------------------------
__global__ __launch_bounds__(128) void out_proj_kernel(
    const _Float16* __restrict__ Oh, const float* __restrict__ Wout,
    const float* __restrict__ bout, float* __restrict__ out) {
  __shared__ _Float16 As[64][32];
  __shared__ _Float16 Bs[64][32];

  const int tid = threadIdx.x;
  const int lane = tid & 31, wave = tid >> 5;
  const int wm = wave >> 1, wn = wave & 1;
  const int n0 = blockIdx.x * 64;
  const int m0 = blockIdx.y * 64;

  const int arow = tid >> 1, ac0 = (tid & 1) * 16;

  v8f acc[2][2] = {};

  for (int kt = 0; kt < kC / 32; ++kt) {
    const int k0 = kt * 32;
    {  // A tile is already f16: async straight into LDS (2 ops/thread)
      const _Float16* src = Oh + (size_t)(m0 + arow) * kC + k0 + ac0;
      async_b128(src,     &As[arow][ac0]);
      async_b128(src + 8, &As[arow][ac0 + 8]);
    }
    stage_w_tile(Wout, kC, k0, n0, Bs, tid);
    wait_asynccnt<0>();
    __syncthreads();

    v16h a0 = frag_a(&As[wm * 32 + 0][0], 32, lane);
    v16h a1 = frag_a(&As[wm * 32 + 16][0], 32, lane);
    v16h b0 = frag_b(&Bs[wn * 32 + 0][0], 32, lane);
    v16h b1 = frag_b(&Bs[wn * 32 + 16][0], 32, lane);
    acc[0][0] = wmma_f16(a0, b0, acc[0][0]);
    acc[0][1] = wmma_f16(a0, b1, acc[0][1]);
    acc[1][0] = wmma_f16(a1, b0, acc[1][0]);
    acc[1][1] = wmma_f16(a1, b1, acc[1][1]);
    __syncthreads();
  }

  const int half8 = (lane >> 4) << 3;
#pragma unroll
  for (int im = 0; im < 2; ++im)
#pragma unroll
    for (int in = 0; in < 2; ++in) {
      const int gn = n0 + wn * 32 + in * 16 + (lane & 15);
      const float bias = bout[gn];
#pragma unroll
      for (int r = 0; r < 8; ++r) {
        const int gm = m0 + wm * 32 + im * 16 + half8 + r;
        out[(size_t)gm * kC + gn] = acc[im][in][r] + bias;
      }
    }
}

// ---------------------------------------------------------------------------
extern "C" void kernel_launch(void* const* d_in, const int* in_sizes, int n_in,
                              void* d_out, int out_size, void* d_ws,
                              size_t ws_size, hipStream_t stream) {
  const float* x    = (const float*)d_in[0];
  const float* Wqkv = (const float*)d_in[1];
  const float* bqkv = (const float*)d_in[2];
  const float* Wout = (const float*)d_in[3];
  const float* bout = (const float*)d_in[4];
  float* out = (float*)d_out;

  // Workspace: f16 Q[B,H,T,D], K[B,H,T,D], V[B,H,D,T], O[B,T,C].
  const size_t HEAD_ELEMS = (size_t)kB * kH * kT * kD;   // 6,291,456
  char* ws = (char*)d_ws;
  _Float16* Qh = (_Float16*)(ws + 0 * HEAD_ELEMS * 2);
  _Float16* Kh = (_Float16*)(ws + 1 * HEAD_ELEMS * 2);
  _Float16* Vh = (_Float16*)(ws + 2 * HEAD_ELEMS * 2);
  _Float16* Oh = (_Float16*)(ws + 3 * HEAD_ELEMS * 2);

  qkv_proj_kernel<<<dim3(k3C / 64, (kB * kT) / 64), 128, 0, stream>>>(
      x, Wqkv, bqkv, Qh, Kh, Vh);
  attn_kernel<<<dim3(kT / 64, kH, kB), 128, 0, stream>>>(Qh, Kh, Vh, Oh);
  out_proj_kernel<<<dim3(kC / 64, (kB * kT) / 64), 128, 0, stream>>>(
      Oh, Wout, bout, out);
}
